// RWKV7TimeMixForTabICL_55250459295926
// MI455X (gfx1250) — matrix operand order repaired and checked by reading, not compiled, not run on hardware
//
#include <hip/hip_runtime.h>
#include <hip/hip_bf16.h>

typedef __bf16 bf16_t;
typedef __bf16 v16bf __attribute__((ext_vector_type(16)));
typedef float  v8f   __attribute__((ext_vector_type(8)));
typedef int    v4i   __attribute__((ext_vector_type(4)));

#define D_MODEL 1024
#define NHEAD   16
#define HEAD_DIM 64
#define BATCH   4
#define SEQ     1024
#define M_TOK   (BATCH*SEQ)   /* 4096 */

#define BM 128
#define BN 128
#define BK 32
#define LDA 40   /* bf16 elements per LDS row: 80B, multiple of 16B, conflict-free */

// gfx1250 async global->LDS path (ASYNCcnt); guarded so compile is safe on any toolchain
#if defined(__has_builtin)
#  if __has_builtin(__builtin_amdgcn_global_load_async_to_lds_b128) && \
      __has_builtin(__builtin_amdgcn_s_wait_asynccnt)
#    define USE_ASYNC_LDS 1
#  endif
#endif
#ifndef USE_ASYNC_LDS
#  define USE_ASYNC_LDS 0
#endif

#if USE_ASYNC_LDS
__device__ __forceinline__ void async_copy_b128(const void* gsrc, void* ldst) {
  __builtin_amdgcn_global_load_async_to_lds_b128(
      (__attribute__((address_space(1))) v4i*)(void*)gsrc,
      (__attribute__((address_space(3))) v4i*)ldst, 0, 0);
}
#endif

struct U8 { uint4 a, b; };

// round-to-nearest-even f32 -> bf16 bits
__device__ __forceinline__ unsigned short f2bf(float f) {
  unsigned u = __builtin_bit_cast(unsigned, f);
  unsigned r = 0x7FFFu + ((u >> 16) & 1u);
  u += r;
  return (unsigned short)(u >> 16);
}
__device__ __forceinline__ unsigned pack_bf16x2(float lo, float hi) {
  return (unsigned)f2bf(lo) | ((unsigned)f2bf(hi) << 16);
}

// ---------------------------------------------------------------------------
// Weight f32 -> bf16 conversion (once; 8 matrices -> 16MB, resident in L2)
// ---------------------------------------------------------------------------
__global__ void __launch_bounds__(256)
cvt_w_kernel(const float* __restrict__ src, bf16_t* __restrict__ dst, int n) {
  int idx = (blockIdx.x * 256 + threadIdx.x) * 4;
  if (idx < n) {
    float4 v = *(const float4*)(src + idx);
    uint2 o;
    o.x = pack_bf16x2(v.x, v.y);
    o.y = pack_bf16x2(v.z, v.w);
    *(uint2*)(dst + idx) = o;
  }
}

// ---------------------------------------------------------------------------
// bf16 WMMA GEMM with double-buffered LDS staging:
//   out[m,n] = sum_k A[m,k] * W[n,k]    (einsum btd,ed->bte)
// mode 0 (MIX):    A[m,k] = x[m,k] + (prev[m,k]-x[m,k])*mix[z][k] -> bf16
// mode 1 (DIRECT): A = adirect (bf16), epilogue adds bias
// grid = (N/BN, M/BM, nz); block = 256 (8 wave32)
// ---------------------------------------------------------------------------
__global__ void __launch_bounds__(256)
wmma_gemm_kernel(const float*  __restrict__ xsrc,
                 const bf16_t* __restrict__ adirect,
                 const bf16_t* __restrict__ wb,
                 const float*  __restrict__ mixcat,
                 float*        __restrict__ outb,
                 const float*  __restrict__ bias,
                 int mode)
{
  __shared__ __align__(16) bf16_t As[2][BM][LDA];
  __shared__ __align__(16) bf16_t Bs[2][BN][LDA];

  const int tid = threadIdx.x;
  const int z   = blockIdx.z;
  const int n0  = blockIdx.x * BN;
  const int m0  = blockIdx.y * BM;

  const bf16_t* wz   = wb + (size_t)z * D_MODEL * D_MODEL;
  const float*  mixz = mixcat ? (mixcat + z * D_MODEL) : (const float*)nullptr;
  float*        outz = outb + (size_t)z * M_TOK * D_MODEL;

  const int wave = tid >> 5;
  const int lane = tid & 31;
  const int wm   = wave >> 2;   // 0..1  -> 64 rows
  const int wn   = wave & 3;    // 0..3  -> 32 cols
  const int half = lane >> 4;
  const int lr   = lane & 15;

  v8f acc[4][2];
  #pragma unroll
  for (int a = 0; a < 4; a++)
    #pragma unroll
    for (int b = 0; b < 2; b++)
      acc[a][b] = {};

  const int srow  = tid >> 1;          // 0..127
  const int skoff = (tid & 1) * 16;    // 0 or 16

  const bf16_t* bsrc = wz + (size_t)(n0 + srow) * D_MODEL + skoff;
  const bf16_t* asrc = (mode == 1)
      ? adirect + (size_t)(m0 + srow) * D_MODEL + skoff : (const bf16_t*)nullptr;
  const int   grow = m0 + srow;
  const int   tpos = grow & (SEQ - 1);
  const float* xrow = (mode == 0) ? xsrc + (size_t)grow * D_MODEL + skoff
                                  : (const float*)nullptr;

  // register staging (sync fallback + MIX-mode A in both paths)
  uint4  rb0, rb1, ra0, ra1;
  float4 rx[4], rp[4], rm[4];

  // ---- stage: issue loads for K-tile k0 into buffer `buf` ----
  auto stage = [&](int buf, int k0) {
#if USE_ASYNC_LDS
    async_copy_b128(bsrc + k0,     &Bs[buf][srow][skoff]);
    async_copy_b128(bsrc + k0 + 8, &Bs[buf][srow][skoff + 8]);
#else
    rb0 = *(const uint4*)(bsrc + k0);
    rb1 = *(const uint4*)(bsrc + k0 + 8);
#endif
    if (k0 + BK < D_MODEL)
      __builtin_prefetch(bsrc + k0 + BK, 0, 1);
    if (mode == 1) {
#if USE_ASYNC_LDS
      async_copy_b128(asrc + k0,     &As[buf][srow][skoff]);
      async_copy_b128(asrc + k0 + 8, &As[buf][srow][skoff + 8]);
#else
      ra0 = *(const uint4*)(asrc + k0);
      ra1 = *(const uint4*)(asrc + k0 + 8);
#endif
    } else {
      #pragma unroll
      for (int q = 0; q < 4; q++) {
        rx[q] = *(const float4*)(xrow + k0 + q * 4);
        if (tpos == 0) rp[q] = make_float4(0.f, 0.f, 0.f, 0.f);
        else           rp[q] = *(const float4*)(xrow - D_MODEL + k0 + q * 4);
        rm[q] = *(const float4*)(mixz + k0 + skoff + q * 4);
      }
    }
  };

  // ---- commit: make buffer `buf` visible in LDS ----
  auto commit = [&](int buf) {
#if USE_ASYNC_LDS
    if (mode == 0) {
      unsigned packed[8];
      #pragma unroll
      for (int q = 0; q < 4; q++) {
        float o0 = rx[q].x + (rp[q].x - rx[q].x) * rm[q].x;
        float o1 = rx[q].y + (rp[q].y - rx[q].y) * rm[q].y;
        float o2 = rx[q].z + (rp[q].z - rx[q].z) * rm[q].z;
        float o3 = rx[q].w + (rp[q].w - rx[q].w) * rm[q].w;
        packed[q * 2 + 0] = pack_bf16x2(o0, o1);
        packed[q * 2 + 1] = pack_bf16x2(o2, o3);
      }
      *(uint4*)(&As[buf][srow][skoff])     = *(uint4*)&packed[0];
      *(uint4*)(&As[buf][srow][skoff + 8]) = *(uint4*)&packed[4];
    }
    __builtin_amdgcn_s_wait_asynccnt(0);
#else
    *(uint4*)(&Bs[buf][srow][skoff])     = rb0;
    *(uint4*)(&Bs[buf][srow][skoff + 8]) = rb1;
    if (mode == 1) {
      *(uint4*)(&As[buf][srow][skoff])     = ra0;
      *(uint4*)(&As[buf][srow][skoff + 8]) = ra1;
    } else {
      unsigned packed[8];
      #pragma unroll
      for (int q = 0; q < 4; q++) {
        float o0 = rx[q].x + (rp[q].x - rx[q].x) * rm[q].x;
        float o1 = rx[q].y + (rp[q].y - rx[q].y) * rm[q].y;
        float o2 = rx[q].z + (rp[q].z - rx[q].z) * rm[q].z;
        float o3 = rx[q].w + (rp[q].w - rx[q].w) * rm[q].w;
        packed[q * 2 + 0] = pack_bf16x2(o0, o1);
        packed[q * 2 + 1] = pack_bf16x2(o2, o3);
      }
      *(uint4*)(&As[buf][srow][skoff])     = *(uint4*)&packed[0];
      *(uint4*)(&As[buf][srow][skoff + 8]) = *(uint4*)&packed[4];
    }
#endif
  };

  int buf = 0;
  stage(0, 0);

  for (int k0 = 0; k0 < D_MODEL; k0 += BK) {
    commit(buf);
    __syncthreads();                       // tile `buf` ready for all waves
    if (k0 + BK < D_MODEL)
      stage(buf ^ 1, k0 + BK);             // overlap next tile with compute

    // ---- compute: 8 x v_wmma_f32_16x16x32_bf16 per wave per K-step ----
    v16bf afrag[4], bfrag[2];
    #pragma unroll
    for (int mt = 0; mt < 4; mt++) {
      const int r = wm * 64 + mt * 16 + lr;              // M = lr
      U8 t;
      t.a = *(const uint4*)(&As[buf][r][half * 8]);      // K = half*8 .. +7
      t.b = *(const uint4*)(&As[buf][r][16 + half * 8]); // K = 16+half*8 .. +7
      afrag[mt] = __builtin_bit_cast(v16bf, t);
    }
    #pragma unroll
    for (int nt = 0; nt < 2; nt++) {
      const int r = wn * 32 + nt * 16 + lr;              // N = lr
      U8 t;
      t.a = *(const uint4*)(&Bs[buf][r][half * 8]);      // K = half*8 .. +7
      t.b = *(const uint4*)(&Bs[buf][r][16 + half * 8]); // K = 16+half*8 .. +7
      bfrag[nt] = __builtin_bit_cast(v16bf, t);
    }
    #pragma unroll
    for (int mt = 0; mt < 4; mt++)
      #pragma unroll
      for (int nt = 0; nt < 2; nt++)
        acc[mt][nt] = __builtin_amdgcn_wmma_f32_16x16x32_bf16(
            false, afrag[mt], false, bfrag[nt], (short)0, acc[mt][nt], false, false);
    __syncthreads();                       // reads of `buf` done before overwrite
    buf ^= 1;
  }

  // ---- epilogue: D layout  M = i + 8*half, N = lr ----
  #pragma unroll
  for (int mt = 0; mt < 4; mt++) {
    #pragma unroll
    for (int nt = 0; nt < 2; nt++) {
      #pragma unroll
      for (int i = 0; i < 8; i++) {
        int gr = m0 + wm * 64 + mt * 16 + half * 8 + i;
        int gc = n0 + wn * 32 + nt * 16 + lr;
        float v = acc[mt][nt][i];
        if (bias) v += bias[gc];
        outz[(size_t)gr * D_MODEL + gc] = v;
      }
    }
  }
}

// ---------------------------------------------------------------------------
// Activations in place on P[z][4096][1024]:
//   z=1: exp(-softplus(raw)) = sigmoid(-raw)   z=2: tanh   z=6: sigmoid
//   z=4, z=5: per-head (64ch) L2 norm: t / max(||t||, 1e-6)
// ---------------------------------------------------------------------------
__global__ void __launch_bounds__(256)
activate_kernel(float* __restrict__ P) {
  const size_t MD = (size_t)M_TOK * D_MODEL;
  const int m = blockIdx.x;
  const int tid = threadIdx.x;
  const size_t rowoff = (size_t)m * D_MODEL + tid * 4;
  __shared__ float red[256];
  __shared__ float nrm[16];

  { // z=1 -> decay factor ew
    float4 v = *(float4*)(P + 1 * MD + rowoff);
    v.x = 1.f / (1.f + __expf(v.x));
    v.y = 1.f / (1.f + __expf(v.y));
    v.z = 1.f / (1.f + __expf(v.z));
    v.w = 1.f / (1.f + __expf(v.w));
    *(float4*)(P + 1 * MD + rowoff) = v;
  }
  { // z=2 -> tanh
    float4 v = *(float4*)(P + 2 * MD + rowoff);
    v.x = tanhf(v.x); v.y = tanhf(v.y); v.z = tanhf(v.z); v.w = tanhf(v.w);
    *(float4*)(P + 2 * MD + rowoff) = v;
  }
  { // z=6 -> sigmoid gate
    float4 v = *(float4*)(P + 6 * MD + rowoff);
    v.x = 1.f / (1.f + __expf(-v.x));
    v.y = 1.f / (1.f + __expf(-v.y));
    v.z = 1.f / (1.f + __expf(-v.z));
    v.w = 1.f / (1.f + __expf(-v.w));
    *(float4*)(P + 6 * MD + rowoff) = v;
  }
  #pragma unroll
  for (int z = 4; z <= 5; z++) { // l2norm for a, b
    float4 v = *(float4*)(P + (size_t)z * MD + rowoff);
    red[tid] = v.x * v.x + v.y * v.y + v.z * v.z + v.w * v.w;
    __syncthreads();
    if (tid < 16) {
      float s = 0.f;
      #pragma unroll
      for (int j = 0; j < 16; j++) s += red[tid * 16 + j];
      nrm[tid] = fmaxf(sqrtf(s), 1e-6f);
    }
    __syncthreads();
    float inv = 1.f / nrm[tid >> 4];
    v.x *= inv; v.y *= inv; v.z *= inv; v.w *= inv;
    *(float4*)(P + (size_t)z * MD + rowoff) = v;
    __syncthreads();
  }
}

// ---------------------------------------------------------------------------
// RWKV7 scan: one WG per (b,h). 256 threads; thread (i = tid&63, gq = tid>>6)
// owns S[i][gq*16 .. gq*16+15] in registers.
//   S = S*ew + (S a) b^T + v k^T ;  o = S r
// ---------------------------------------------------------------------------
__global__ void __launch_bounds__(256)
scan_kernel(const float* __restrict__ P, float* __restrict__ O) {
  const size_t MD = (size_t)M_TOK * D_MODEL;
  const int wg = blockIdx.x;          // 0..63
  const int bb = wg >> 4, hh = wg & 15;
  const int tid = threadIdx.x;
  const int i  = tid & 63;
  const int gq = tid >> 6;            // 0..3 column group

  float s[16];
  #pragma unroll
  for (int j = 0; j < 16; j++) s[j] = 0.f;

  __shared__ float vec[6 * 64];       // r, ew, k, v, a, b
  __shared__ float red[4][64];
  __shared__ float red2[4][64];

  for (int t = 0; t < SEQ; t++) {
    const size_t base = ((size_t)(bb * SEQ + t)) * D_MODEL + hh * HEAD_DIM;
    for (int j = tid; j < 384; j += 256) {
      int arr = j >> 6, c = j & 63;   // arr: 0=r 1=ew 2=k 3=v 4=a 5=b
      vec[j] = P[(size_t)arr * MD + base + c];
    }
    __syncthreads();
    const float* rr = vec;
    const float* ew = vec + 64;
    const float* kk = vec + 128;
    const float* vv = vec + 192;
    const float* aa = vec + 256;
    const float* bv = vec + 320;

    float p = 0.f;
    #pragma unroll
    for (int j = 0; j < 16; j++) p += s[j] * aa[gq * 16 + j];
    red[gq][i] = p;
    __syncthreads();
    const float Sa = red[0][i] + red[1][i] + red[2][i] + red[3][i];
    const float vi = vv[i];

    float q = 0.f;
    #pragma unroll
    for (int j = 0; j < 16; j++) {
      const int c = gq * 16 + j;
      s[j] = s[j] * ew[c] + Sa * bv[c] + vi * kk[c];
      q += s[j] * rr[c];
    }
    red2[gq][i] = q;
    __syncthreads();
    if (gq == 0)
      O[base + i] = red2[0][i] + red2[1][i] + red2[2][i] + red2[3][i];
    __syncthreads();
  }
}

// ---------------------------------------------------------------------------
// GroupNorm(16 groups, eps=0.00064) + scale/shift + sigmoid gate -> bf16 A
// ---------------------------------------------------------------------------
__global__ void __launch_bounds__(256)
gnorm_gate_kernel(const float* __restrict__ O, const float* __restrict__ G,
                  const float* __restrict__ gnw, const float* __restrict__ gnb,
                  bf16_t* __restrict__ NG) {
  const int m = blockIdx.x;
  const int tid = threadIdx.x;
  const int ch0 = tid * 4;
  const size_t rowoff = (size_t)m * D_MODEL + ch0;

  float4 v = *(const float4*)(O + rowoff);
  __shared__ float rs[256], rss[256];
  __shared__ float mu[16], rstd[16];
  rs[tid]  = v.x + v.y + v.z + v.w;
  rss[tid] = v.x * v.x + v.y * v.y + v.z * v.z + v.w * v.w;
  __syncthreads();
  if (tid < 16) {
    float s = 0.f, ss = 0.f;
    #pragma unroll
    for (int j = 0; j < 16; j++) { s += rs[tid * 16 + j]; ss += rss[tid * 16 + j]; }
    float mean = s * (1.f / 64.f);
    float var  = ss * (1.f / 64.f) - mean * mean;
    mu[tid] = mean;
    rstd[tid] = rsqrtf(var + 0.00064f);
  }
  __syncthreads();
  const int h = tid >> 4;
  float4 w4 = *(const float4*)(gnw + ch0);
  float4 b4 = *(const float4*)(gnb + ch0);
  float4 g4 = *(const float4*)(G + rowoff);
  float y0 = ((v.x - mu[h]) * rstd[h] * w4.x + b4.x) * g4.x;
  float y1 = ((v.y - mu[h]) * rstd[h] * w4.y + b4.y) * g4.y;
  float y2 = ((v.z - mu[h]) * rstd[h] * w4.z + b4.z) * g4.z;
  float y3 = ((v.w - mu[h]) * rstd[h] * w4.w + b4.w) * g4.w;
  uint2 o;
  o.x = pack_bf16x2(y0, y1);
  o.y = pack_bf16x2(y2, y3);
  *(uint2*)(NG + rowoff) = o;
}

// ---------------------------------------------------------------------------
// Workspace layout (bytes):
//   Wb   @ 0          : 8 * 1M bf16  = 16,777,216
//   mix  @ 16777216   : 7 * 1024 f32 = 28,672
//   P    @ 16805888   : 7 * 4M f32   = 117,440,512
//   O    @ 134246400  : 4M f32       = 16,777,216
//   NG   @ 151023616  : 4M bf16      = 8,388,608     (end ~159.4 MB)
// ---------------------------------------------------------------------------
extern "C" void kernel_launch(void* const* d_in, const int* in_sizes, int n_in,
                              void* d_out, int out_size, void* d_ws, size_t ws_size,
                              hipStream_t stream) {
  (void)in_sizes; (void)n_in; (void)out_size; (void)ws_size;
  const float* x   = (const float*)d_in[0];
  const float* b_o = (const float*)d_in[16];
  const float* gnw = (const float*)d_in[17];
  const float* gnb = (const float*)d_in[18];

  char*   ws  = (char*)d_ws;
  bf16_t* Wb  = (bf16_t*)(ws);
  float*  mix = (float*)(ws + 16777216);
  float*  P   = (float*)(ws + 16805888);
  float*  O   = (float*)(ws + 134246400);
  bf16_t* NG  = (bf16_t*)(ws + 151023616);

  // weights f32 -> bf16 : order r,w,k,v,a,b,g,o (d_in[8..15])
  for (int i = 0; i < 8; i++)
    cvt_w_kernel<<<dim3(1024), dim3(256), 0, stream>>>(
        (const float*)d_in[8 + i], Wb + (size_t)i * 1048576, 1048576);

  // token-shift mix vectors (r,w,k,v,a,b,g = d_in[1..7])
  for (int i = 0; i < 7; i++)
    (void)hipMemcpyAsync(mix + i * 1024, d_in[1 + i], 1024 * sizeof(float),
                         hipMemcpyDeviceToDevice, stream);

  // 7 fused mix+projection GEMMs (z = projection index)
  wmma_gemm_kernel<<<dim3(8, 32, 7), dim3(256), 0, stream>>>(
      x, (const bf16_t*)nullptr, Wb, mix, P, (const float*)nullptr, 0);

  // activations (decay, tanh, sigmoid, L2 norms)
  activate_kernel<<<dim3(4096), dim3(256), 0, stream>>>(P);

  // sequential RWKV7 delta-rule scan
  scan_kernel<<<dim3(64), dim3(256), 0, stream>>>(P, O);

  // groupnorm + gate -> bf16 activations for output projection
  gnorm_gate_kernel<<<dim3(4096), dim3(256), 0, stream>>>(
      O, P + 6 * (size_t)M_TOK * D_MODEL, gnw, gnb, NG);

  // output GEMM with bias epilogue (W_o = Wb slot 7)
  wmma_gemm_kernel<<<dim3(8, 32, 1), dim3(256), 0, stream>>>(
      (const float*)nullptr, NG, Wb + 7 * (size_t)1048576,
      (const float*)nullptr, (float*)d_out, b_o, 1);
}
